// RBPXFormer_62646392980071
// MI455X (gfx1250) — matrix operand order repaired and checked by reading, not compiled, not run on hardware
//
#include <hip/hip_runtime.h>

// ---------------------------------------------------------------------------
// Types
// ---------------------------------------------------------------------------
typedef _Float16 v8h  __attribute__((ext_vector_type(8)));
typedef _Float16 v16h __attribute__((ext_vector_type(16)));
typedef float    v8f  __attribute__((ext_vector_type(8)));

#define BM 128
#define BN 64
#define BK 32
#define LDT 40   // padded halfs per LDS tile row -> conflict-free ds_load_b128

// Model dims
#define B_   4
#define N1_  1024
#define N2_  768
#define DIM_ 1536
#define H_   8
#define DK_  64
#define DV_  192
#define HDK_ 512
#define HDV_ 1536
#define L_   2047
#define FFN_ 3072

// ---------------------------------------------------------------------------
// CDNA5 async global->LDS copy (16B per lane) + completion wait
// ---------------------------------------------------------------------------
__device__ __forceinline__ void async_load_b128(const _Float16* gaddr,
                                                unsigned lds_off) {
  asm volatile("global_load_async_to_lds_b128 %0, %1, off"
               :: "v"(lds_off), "v"((unsigned long long)(size_t)gaddr)
               : "memory");
}
__device__ __forceinline__ void wait_async0() {
  asm volatile("s_wait_asynccnt 0" ::: "memory");
}

// ---------------------------------------------------------------------------
// Generic batched WMMA GEMM, all-f16 operands, k-contiguous A and B.
// A elem: A + bb*sAb + hh*sAh + m*sAm + k      (f16)
// B elem: B + bb*sBb + hh*sBh + n*sBn + k      (f16)
// C elem: C + bb*sCb + hh*sCh + m*sCm + n      (f32, or f16 when flags&8)
// Residual R (f32) uses C strides. K must be a multiple of 32.
// flags: 1=bias  2=relu  4=residual  8=store f16
// ---------------------------------------------------------------------------
struct GemmP {
  const _Float16* A; const _Float16* B; void* C;
  const float* bias; const float* R;
  long sAm, sAb, sAh;
  long sBn, sBb, sBh;
  long sCm, sCb, sCh;
  int  M, N, K, Hd, flags;
  float scale;
};

__global__ __launch_bounds__(256) void gemm_wmma_f16(GemmP p) {
  __shared__ __align__(16) _Float16 aT[BM * LDT];
  __shared__ __align__(16) _Float16 bT[BN * LDT];

  const int z  = blockIdx.z;
  const int bb = z / p.Hd, hh = z % p.Hd;
  const _Float16* Ab = p.A + (long)bb * p.sAb + (long)hh * p.sAh;
  const _Float16* Bb = p.B + (long)bb * p.sBb + (long)hh * p.sBh;

  const int m0 = blockIdx.y * BM, n0 = blockIdx.x * BN;
  const int t = threadIdx.x, lane = t & 31, wave = t >> 5;
  const int wm = (wave & 3) * 32, wn = (wave >> 2) * 32;
  const int fm = lane & 15, half = lane >> 4;

  // Branch-free staging assignments (one 16-half chunk per thread).
  // Row indices clamped: out-of-range rows/cols only feed skipped C elements.
  const int  ar  = t >> 1, ac = (t & 1) * 16;
  const long agm = (m0 + ar < p.M) ? (long)(m0 + ar) : (long)(p.M - 1);
  const unsigned aDst = (unsigned)(size_t)&aT[ar * LDT + ac];
  const int  br  = (t & 127) >> 1, bc = (t & 1) * 16;
  const long bgn = (n0 + br < p.N) ? (long)(n0 + br) : (long)(p.N - 1);
  const unsigned bDst = (unsigned)(size_t)&bT[br * LDT + bc];
  const bool doB = (t < 128);                 // wave-uniform

  const _Float16* aSrc = Ab + agm * p.sAm + ac;
  const _Float16* bSrc = Bb + bgn * p.sBn + bc;

  v8f acc[2][2] = {};

  for (int k0 = 0; k0 < p.K; k0 += BK) {
    async_load_b128(aSrc + k0, aDst);
    if (doB) async_load_b128(bSrc + k0, bDst);
    wait_async0();
    __syncthreads();

    v16h afr[2], bfr[2];
    #pragma unroll
    for (int i = 0; i < 2; ++i) {
      // A 16x32 layout: halfs[0..7]=K(half*8..), halfs[8..15]=K(16+half*8..)
      const _Float16* pa = &aT[(wm + i * 16 + fm) * LDT + half * 8];
      v8h alo = *(const v8h*)pa;
      v8h ahi = *(const v8h*)(pa + 16);
      afr[i] = __builtin_shufflevector(alo, ahi, 0, 1, 2, 3, 4, 5, 6, 7, 8, 9,
                                       10, 11, 12, 13, 14, 15);
      // B 32x16 layout: halfs[0..7]=K(half*8..), halfs[8..15]=K(16+half*8..)
      const _Float16* pb = &bT[(wn + i * 16 + fm) * LDT + half * 8];
      v8h blo = *(const v8h*)pb;
      v8h bhi = *(const v8h*)(pb + 16);
      bfr[i] = __builtin_shufflevector(blo, bhi, 0, 1, 2, 3, 4, 5, 6, 7, 8, 9,
                                       10, 11, 12, 13, 14, 15);
    }
    #pragma unroll
    for (int i = 0; i < 2; ++i)
      #pragma unroll
      for (int j = 0; j < 2; ++j)
        acc[i][j] = __builtin_amdgcn_wmma_f32_16x16x32_f16(
            false, afr[i], false, bfr[j], (short)0, acc[i][j], false, false);
    __syncthreads();
  }

  // Epilogue: C/D layout -> row = (lane>>4)*8 + v, col = lane&15
  float*     Cf = (float*)p.C + (long)bb * p.sCb + (long)hh * p.sCh;
  _Float16*  Ch = (_Float16*)p.C + (long)bb * p.sCb + (long)hh * p.sCh;
  const float* Rb = (p.flags & 4)
                        ? (p.R + (long)bb * p.sCb + (long)hh * p.sCh) : nullptr;
  #pragma unroll
  for (int i = 0; i < 2; ++i) {
    #pragma unroll
    for (int j = 0; j < 2; ++j) {
      const int col = n0 + wn + j * 16 + fm;
      if (col >= p.N) continue;
      const int rbase = m0 + wm + i * 16 + half * 8;
      #pragma unroll
      for (int v = 0; v < 8; ++v) {
        const int row = rbase + v;
        if (row >= p.M) continue;
        float r = acc[i][j][v] * p.scale;
        if (p.flags & 1) r += p.bias[col];
        if (p.flags & 4) r += Rb[(long)row * p.sCm + col];
        if (p.flags & 2) r = fmaxf(r, 0.f);
        if (p.flags & 8) Ch[(long)row * p.sCm + col] = (_Float16)r;
        else             Cf[(long)row * p.sCm + col] = r;
      }
    }
  }
}

// ---------------------------------------------------------------------------
// fp32 -> f16 contiguous convert (8 elems/thread)
// ---------------------------------------------------------------------------
__global__ __launch_bounds__(256) void cvt_f16(const float* in, _Float16* out,
                                               long n) {
  const long i = ((long)blockIdx.x * 256 + threadIdx.x) * 8;
  if (i + 8 <= n) {
    const float4 a = *(const float4*)(in + i);
    const float4 b = *(const float4*)(in + i + 4);
    v8h h = {(_Float16)a.x, (_Float16)a.y, (_Float16)a.z, (_Float16)a.w,
             (_Float16)b.x, (_Float16)b.y, (_Float16)b.z, (_Float16)b.w};
    *(v8h*)(out + i) = h;
  }
}

// ---------------------------------------------------------------------------
// fp32 [R,C] -> f16 [C,R] transpose-convert (R,C multiples of 32)
// ---------------------------------------------------------------------------
__global__ __launch_bounds__(256) void cvtT_f16(const float* in, _Float16* out,
                                                int R, int C) {
  __shared__ float tile[32][33];
  const int c0 = blockIdx.x * 32, r0 = blockIdx.y * 32;
  const int t = threadIdx.x, tc = t & 31, tr = t >> 5;
  #pragma unroll
  for (int i = 0; i < 4; ++i)
    tile[tr + i * 8][tc] = in[(long)(r0 + tr + i * 8) * C + c0 + tc];
  __syncthreads();
  #pragma unroll
  for (int i = 0; i < 4; ++i)
    out[(long)(c0 + tr + i * 8) * R + r0 + tc] = (_Float16)tile[tc][tr + i * 8];
}

// ---------------------------------------------------------------------------
// Batched f16 [R,C] -> f16 [z][C,R] transpose (strided input rows)
// in elem: in + bb*inZb + hh*inZh + r*sRow + c ; out elem: out + z*C*R + c*R + r
// ---------------------------------------------------------------------------
struct TransP {
  const _Float16* in; _Float16* out;
  long inZb, inZh, sRow;
  int R, C, Hd;
};
__global__ __launch_bounds__(256) void transpose_f16(TransP tp) {
  __shared__ _Float16 tile[32][33];
  const int z = blockIdx.z, bb = z / tp.Hd, hh = z % tp.Hd;
  const _Float16* ib = tp.in + (long)bb * tp.inZb + (long)hh * tp.inZh;
  _Float16* ob = tp.out + (long)z * tp.C * tp.R;
  const int r0 = blockIdx.y * 32, c0 = blockIdx.x * 32;
  const int t = threadIdx.x, tc = t & 31, tr = t >> 5;
  #pragma unroll
  for (int i = 0; i < 4; ++i)
    tile[tr + i * 8][tc] = ib[(long)(r0 + tr + i * 8) * tp.sRow + c0 + tc];
  __syncthreads();
  #pragma unroll
  for (int i = 0; i < 4; ++i)
    ob[(long)(c0 + tr + i * 8) * tp.R + r0 + tc] = tile[tc][tr + i * 8];
}

// ---------------------------------------------------------------------------
// Enformer positional features -> f16 pos[2047,192]. One wave per row.
// ---------------------------------------------------------------------------
__global__ __launch_bounds__(32) void posembed_kernel(_Float16* pos) {
  const int r = blockIdx.x, t = threadIdx.x;
  const float dist = (float)(r - (N1_ - 1));
  const float ad = fabsf(dist);
  const float hl = exp2f(3.0f + (float)t * (7.0f / 31.0f));
  const float fe = exp2f(-ad / hl);
  const float w = exp2f((float)(t + 1)) - 1.0f;
  const float fc = (w > ad) ? 1.0f : 0.0f;
  const float mean = 32.0f * (float)(t + 1);
  const float conc = (mean / 16.0f) * (mean / 16.0f);
  const float rate = mean / 256.0f;
  const float logp = ((ad == 0.0f) ? 0.0f : (conc - 1.0f) * logf(ad)) - rate * ad;
  const float logn = lgammaf(conc) - conc * logf(rate);
  const float prob = expf(logp - logn) + 1e-8f;
  float mx = prob;
  #pragma unroll
  for (int s = 16; s > 0; s >>= 1) mx = fmaxf(mx, __shfl_xor(mx, s, 32));
  const float fg = prob / mx;
  const float sg = (dist > 0.f) ? 1.f : ((dist < 0.f) ? -1.f : 0.f);
  _Float16* row = pos + (long)r * 192;
  row[t]      = (_Float16)fe;        row[32 + t]  = (_Float16)fc;
  row[64 + t] = (_Float16)fg;        row[96 + t]  = (_Float16)(sg * fe);
  row[128 + t] = (_Float16)(sg * fc); row[160 + t] = (_Float16)(sg * fg);
}

// ---------------------------------------------------------------------------
// LayerNorm over 1536 cols (f32 in, f16 out), one block per row
// ---------------------------------------------------------------------------
__global__ __launch_bounds__(256) void layernorm_1536(const float* in,
                                                      const float* g,
                                                      const float* bta,
                                                      _Float16* out) {
  const long row = blockIdx.x;
  const float* p = in + row * DIM_;
  const int t = threadIdx.x;
  __shared__ float red[256];
  float v[6]; float s = 0.f;
  #pragma unroll
  for (int i = 0; i < 6; ++i) { v[i] = p[t + i * 256]; s += v[i]; }
  red[t] = s; __syncthreads();
  for (int k = 128; k > 0; k >>= 1) { if (t < k) red[t] += red[t + k]; __syncthreads(); }
  const float mu = red[0] * (1.f / (float)DIM_);
  __syncthreads();
  float s2 = 0.f;
  #pragma unroll
  for (int i = 0; i < 6; ++i) { const float d = v[i] - mu; s2 += d * d; }
  red[t] = s2; __syncthreads();
  for (int k = 128; k > 0; k >>= 1) { if (t < k) red[t] += red[t + k]; __syncthreads(); }
  const float rinv = rsqrtf(red[0] * (1.f / (float)DIM_) + 1e-5f);
  _Float16* o = out + row * DIM_;
  #pragma unroll
  for (int i = 0; i < 6; ++i) {
    const int c = t + i * 256;
    o[c] = (_Float16)((v[i] - mu) * rinv * g[c] + bta[c]);
  }
}

// kb = k*scale + rel_pos_bias (broadcast over rows; last dim = h*64+d)
__global__ void kb_kernel(const _Float16* kin, const float* rpb, _Float16* kb,
                          int n) {
  const int i = blockIdx.x * 256 + threadIdx.x;
  if (i < n) kb[i] = (_Float16)((float)kin[i] * 0.125f + rpb[i & (HDK_ - 1)]);
}

// ---------------------------------------------------------------------------
// Fused relative-shift gather + softmax over j: attn16[z,i,j] (f16 out)
// logits = content[z,i,j] + rel[z,j,i-j+767]
// ---------------------------------------------------------------------------
__global__ __launch_bounds__(256) void softmax_rel(const float* content,
                                                   const float* rel,
                                                   _Float16* attn) {
  const int i = blockIdx.x, z = blockIdx.y;
  const float* row = content + ((long)z * N1_ + i) * N2_;
  const float* rz = rel + (long)z * N2_ * L_;
  _Float16* orow = attn + ((long)z * N1_ + i) * N2_;
  const int t = threadIdx.x;
  __shared__ float red[256];
  float v[3];
  #pragma unroll
  for (int s = 0; s < 3; ++s) {
    const int j = t + s * 256;
    v[s] = row[j] + rz[(long)j * L_ + (i - j + (N1_ - 257))];
  }
  float m = fmaxf(v[0], fmaxf(v[1], v[2]));
  red[t] = m; __syncthreads();
  for (int k = 128; k > 0; k >>= 1) { if (t < k) red[t] = fmaxf(red[t], red[t + k]); __syncthreads(); }
  m = red[0]; __syncthreads();
  const float e0 = __expf(v[0] - m), e1 = __expf(v[1] - m), e2 = __expf(v[2] - m);
  red[t] = e0 + e1 + e2; __syncthreads();
  for (int k = 128; k > 0; k >>= 1) { if (t < k) red[t] += red[t + k]; __syncthreads(); }
  const float inv = 1.f / red[0];
  orow[t] = (_Float16)(e0 * inv);
  orow[t + 256] = (_Float16)(e1 * inv);
  orow[t + 512] = (_Float16)(e2 * inv);
}

// ---------------------------------------------------------------------------
// Host launcher
// ---------------------------------------------------------------------------
extern "C" void kernel_launch(void* const* d_in, const int* in_sizes, int n_in,
                              void* d_out, int out_size, void* d_ws,
                              size_t ws_size, hipStream_t stream) {
  (void)in_sizes; (void)n_in; (void)out_size; (void)ws_size;
  const float* x     = (const float*)d_in[0];
  const float* y0    = (const float*)d_in[1];
  const float* res_w = (const float*)d_in[2];
  const float* lnx_g = (const float*)d_in[3];
  const float* lnx_b = (const float*)d_in[4];
  const float* lny_g = (const float*)d_in[5];
  const float* lny_b = (const float*)d_in[6];
  const float* wq    = (const float*)d_in[7];
  const float* wk    = (const float*)d_in[8];
  const float* wv1   = (const float*)d_in[9];
  const float* wv2   = (const float*)d_in[10];
  const float* wrel  = (const float*)d_in[11];
  const float* rpb   = (const float*)d_in[12];
  const float* wo1   = (const float*)d_in[13];
  const float* bo1   = (const float*)d_in[14];
  const float* wo2   = (const float*)d_in[15];
  const float* bo2   = (const float*)d_in[16];
  const float* fxg   = (const float*)d_in[17];
  const float* fxb   = (const float*)d_in[18];
  const float* fxw1  = (const float*)d_in[19];
  const float* fxb1  = (const float*)d_in[20];
  const float* fxw2  = (const float*)d_in[21];
  const float* fxb2  = (const float*)d_in[22];
  const float* fyg   = (const float*)d_in[23];
  const float* fyb   = (const float*)d_in[24];
  const float* fyw1  = (const float*)d_in[25];
  const float* fyb1  = (const float*)d_in[26];
  const float* fyw2  = (const float*)d_in[27];
  const float* fyb2  = (const float*)d_in[28];

  const size_t MX = (size_t)B_ * N1_;  // 4096
  const size_t MY = (size_t)B_ * N2_;  // 3072
  char* base = (char*)d_ws;
  size_t off = 0;
  auto alloc = [&](size_t bytes) {
    char* p = base + off;
    off = (off + bytes + 255) & ~(size_t)255;
    return p;
  };
  auto fa  = [&](size_t n) { return (float*)alloc(n * 4); };
  auto ha  = [&](size_t n) { return (_Float16*)alloc(n * 2); };

  _Float16* pos16  = ha((size_t)L_ * 192);
  _Float16* relq16 = ha((size_t)L_ * HDK_);
  _Float16* y0h    = ha(MY * DIM_);
  _Float16* reswT  = ha((size_t)DIM_ * DIM_);
  _Float16* wqT    = ha((size_t)HDK_ * DIM_);
  _Float16* wkT    = ha((size_t)HDK_ * DIM_);
  _Float16* wv1T   = ha((size_t)HDV_ * DIM_);
  _Float16* wv2T   = ha((size_t)HDV_ * DIM_);
  _Float16* wo1T   = ha((size_t)DIM_ * HDV_);
  _Float16* wo2T   = ha((size_t)DIM_ * HDV_);
  _Float16* fxw1T  = ha((size_t)FFN_ * DIM_);
  _Float16* fxw2T  = ha((size_t)DIM_ * FFN_);
  _Float16* fyw1T  = ha((size_t)FFN_ * DIM_);
  _Float16* fyw2T  = ha((size_t)DIM_ * FFN_);
  _Float16* wrelT  = ha((size_t)HDK_ * 192);
  float*    yres   = fa(MY * DIM_);
  _Float16* x1h    = ha(MX * DIM_);
  _Float16* y1h    = ha(MY * DIM_);
  _Float16* q16    = ha(MX * HDK_);
  _Float16* k16    = ha(MY * HDK_);
  _Float16* kb16   = ha(MY * HDK_);
  _Float16* v1h    = ha(MX * HDV_);
  _Float16* v2h    = ha(MY * HDV_);
  _Float16* v1T    = ha(MX * HDV_);
  _Float16* v2T    = ha(MY * HDV_);
  _Float16* attn16 = ha((size_t)B_ * H_ * N1_ * N2_);
  float*    content = fa((size_t)B_ * H_ * N1_ * N2_);   // region reused below
  float*    rel     = fa((size_t)B_ * H_ * N2_ * L_);    // region reused below
  // reuse rel region (dead after softmax): attnT + FFN hiddens
  _Float16* attnT = (_Float16*)rel;
  _Float16* hx16  = attnT + (size_t)B_ * H_ * N1_ * N2_;
  _Float16* hy16  = hx16 + MX * (size_t)FFN_;
  // reuse content region (dead after softmax): out1/out2/LN outs/x4/y4
  _Float16* out1h = (_Float16*)content;
  _Float16* out2h = out1h + MX * DIM_;
  _Float16* hxln  = out2h + MY * DIM_;
  _Float16* hyln  = hxln + MX * DIM_;
  float*    x4    = (float*)(hyln + MY * DIM_);
  float*    y4    = x4 + MX * DIM_;
  float*    x5    = (float*)d_out;
  float*    y5    = (float*)d_out + MX * DIM_;

  // ---- operand preparation (convert / transpose to f16) --------------------
  cvt_f16<<<dim3((unsigned)((MY * DIM_ / 8 + 255) / 256)), 256, 0, stream>>>(
      y0, y0h, (long)(MY * DIM_));
  cvtT_f16<<<dim3(DIM_ / 32, DIM_ / 32), 256, 0, stream>>>(res_w, reswT, DIM_, DIM_);
  cvtT_f16<<<dim3(HDK_ / 32, DIM_ / 32), 256, 0, stream>>>(wq, wqT, DIM_, HDK_);
  cvtT_f16<<<dim3(HDK_ / 32, DIM_ / 32), 256, 0, stream>>>(wk, wkT, DIM_, HDK_);
  cvtT_f16<<<dim3(HDV_ / 32, DIM_ / 32), 256, 0, stream>>>(wv1, wv1T, DIM_, HDV_);
  cvtT_f16<<<dim3(HDV_ / 32, DIM_ / 32), 256, 0, stream>>>(wv2, wv2T, DIM_, HDV_);
  cvtT_f16<<<dim3(DIM_ / 32, HDV_ / 32), 256, 0, stream>>>(wo1, wo1T, HDV_, DIM_);
  cvtT_f16<<<dim3(DIM_ / 32, HDV_ / 32), 256, 0, stream>>>(wo2, wo2T, HDV_, DIM_);
  cvtT_f16<<<dim3(FFN_ / 32, DIM_ / 32), 256, 0, stream>>>(fxw1, fxw1T, DIM_, FFN_);
  cvtT_f16<<<dim3(DIM_ / 32, FFN_ / 32), 256, 0, stream>>>(fxw2, fxw2T, FFN_, DIM_);
  cvtT_f16<<<dim3(FFN_ / 32, DIM_ / 32), 256, 0, stream>>>(fyw1, fyw1T, DIM_, FFN_);
  cvtT_f16<<<dim3(DIM_ / 32, FFN_ / 32), 256, 0, stream>>>(fyw2, fyw2T, FFN_, DIM_);
  cvtT_f16<<<dim3(HDK_ / 32, 192 / 32), 256, 0, stream>>>(wrel, wrelT, 192, HDK_);
  posembed_kernel<<<dim3(L_), dim3(32), 0, stream>>>(pos16);

  GemmP p{};
  auto G = [&](const _Float16* A, const _Float16* B, void* C,
               const float* bias, const float* R,
               long sAm, long sAb, long sAh, long sBn, long sBb, long sBh,
               long sCm, long sCb, long sCh,
               int M, int N, int K, int Hd, int nz, int flags, float scale) {
    p.A = A; p.B = B; p.C = C; p.bias = bias; p.R = R;
    p.sAm = sAm; p.sAb = sAb; p.sAh = sAh;
    p.sBn = sBn; p.sBb = sBb; p.sBh = sBh;
    p.sCm = sCm; p.sCb = sCb; p.sCh = sCh;
    p.M = M; p.N = N; p.K = K; p.Hd = Hd; p.flags = flags; p.scale = scale;
    gemm_wmma_f16<<<dim3((N + BN - 1) / BN, (M + BM - 1) / BM, nz), 256, 0,
                    stream>>>(p);
  };

  // relq = pos @ wrel  -> f16 [2047,512]
  G(pos16, wrelT, relq16, nullptr, nullptr, 192, 0, 0, 192, 0, 0,
    HDK_, 0, 0, L_, HDK_, 192, 1, 1, 8, 1.f);
  // y = y0 @ res_w -> f32
  G(y0h, reswT, yres, nullptr, nullptr, DIM_, 0, 0, DIM_, 0, 0,
    DIM_, 0, 0, (int)MY, DIM_, DIM_, 1, 1, 0, 1.f);

  layernorm_1536<<<dim3((unsigned)MX), 256, 0, stream>>>(x, lnx_g, lnx_b, x1h);
  layernorm_1536<<<dim3((unsigned)MY), 256, 0, stream>>>(yres, lny_g, lny_b, y1h);

  // q = (x1@wq)*dk^-0.5 -> f16 ; k = y1@wk -> f16
  G(x1h, wqT, q16, nullptr, nullptr, DIM_, 0, 0, DIM_, 0, 0,
    HDK_, 0, 0, (int)MX, HDK_, DIM_, 1, 1, 8, 0.125f);
  G(y1h, wkT, k16, nullptr, nullptr, DIM_, 0, 0, DIM_, 0, 0,
    HDK_, 0, 0, (int)MY, HDK_, DIM_, 1, 1, 8, 1.f);
  // v2 = y1@wv2 -> f16 ; v1 = x1@wv1 -> f16
  G(y1h, wv2T, v2h, nullptr, nullptr, DIM_, 0, 0, DIM_, 0, 0,
    HDV_, 0, 0, (int)MY, HDV_, DIM_, 1, 1, 8, 1.f);
  G(x1h, wv1T, v1h, nullptr, nullptr, DIM_, 0, 0, DIM_, 0, 0,
    HDV_, 0, 0, (int)MX, HDV_, DIM_, 1, 1, 8, 1.f);

  kb_kernel<<<dim3((unsigned)((MY * HDK_ + 255) / 256)), 256, 0, stream>>>(
      k16, rpb, kb16, (int)(MY * HDK_));

  // per-(b,h) transposes: v2T [z][dv][j], v1T [z][dv][i]
  {
    TransP tp{v2h, v2T, (long)N2_ * HDV_, DV_, HDV_, N2_, DV_, H_};
    transpose_f16<<<dim3(DV_ / 32, N2_ / 32, B_ * H_), 256, 0, stream>>>(tp);
  }
  {
    TransP tp{v1h, v1T, (long)N1_ * HDV_, DV_, HDV_, N1_, DV_, H_};
    transpose_f16<<<dim3(DV_ / 32, N1_ / 32, B_ * H_), 256, 0, stream>>>(tp);
  }

  // content[b,h] = q @ k^T  -> f32 [z,1024,768]
  G(q16, k16, content, nullptr, nullptr,
    HDK_, (long)N1_ * HDK_, DK_, HDK_, (long)N2_ * HDK_, DK_,
    N2_, (long)H_ * N1_ * N2_, (long)N1_ * N2_,
    N1_, N2_, DK_, H_, B_ * H_, 0, 1.f);
  // rel_logits[b,h] = kb @ relq^T -> f32 [z,768,2047]
  G(kb16, relq16, rel, nullptr, nullptr,
    HDK_, (long)N2_ * HDK_, DK_, HDK_, 0, DK_,
    L_, (long)H_ * N2_ * L_, (long)N2_ * L_,
    N2_, L_, DK_, H_, B_ * H_, 0, 1.f);

  // fused rel-shift + softmax -> attn16 (f16)
  softmax_rel<<<dim3(N1_, B_ * H_), 256, 0, stream>>>(content, rel, attn16);

  // attnT [z][j][i]
  {
    TransP tp{attn16, attnT, (long)H_ * N1_ * N2_, (long)N1_ * N2_, N2_,
              N1_, N2_, H_};
    transpose_f16<<<dim3(N2_ / 32, N1_ / 32, B_ * H_), 256, 0, stream>>>(tp);
  }

  // out1[b,h] = attn @ v2 -> f16 [b,i,h,dv]
  G(attn16, v2T, out1h, nullptr, nullptr,
    N2_, (long)H_ * N1_ * N2_, (long)N1_ * N2_,
    N2_, (long)H_ * DV_ * N2_, (long)DV_ * N2_,
    HDV_, (long)N1_ * HDV_, DV_,
    N1_, DV_, N2_, H_, B_ * H_, 8, 1.f);
  // out2[b,h] = attn^T @ v1 -> f16 [b,j,h,dv]
  G(attnT, v1T, out2h, nullptr, nullptr,
    N1_, (long)H_ * N2_ * N1_, (long)N2_ * N1_,
    N1_, (long)H_ * DV_ * N1_, (long)DV_ * N1_,
    HDV_, (long)N2_ * HDV_, DV_,
    N2_, DV_, N1_, H_, B_ * H_, 8, 1.f);

  // x4 = x + out1@wo1 + bo1 ; y4 = y + out2@wo2 + bo2  (f32)
  G(out1h, wo1T, x4, bo1, x, HDV_, 0, 0, HDV_, 0, 0,
    DIM_, 0, 0, (int)MX, DIM_, HDV_, 1, 1, 1 | 4, 1.f);
  G(out2h, wo2T, y4, bo2, yres, HDV_, 0, 0, HDV_, 0, 0,
    DIM_, 0, 0, (int)MY, DIM_, HDV_, 1, 1, 1 | 4, 1.f);

  // x FFN
  layernorm_1536<<<dim3((unsigned)MX), 256, 0, stream>>>(x4, fxg, fxb, hxln);
  G(hxln, fxw1T, hx16, fxb1, nullptr, DIM_, 0, 0, DIM_, 0, 0,
    FFN_, 0, 0, (int)MX, FFN_, DIM_, 1, 1, 1 | 2 | 8, 1.f);
  G(hx16, fxw2T, x5, fxb2, x4, FFN_, 0, 0, FFN_, 0, 0,
    DIM_, 0, 0, (int)MX, DIM_, FFN_, 1, 1, 1 | 4, 1.f);

  // y FFN
  layernorm_1536<<<dim3((unsigned)MY), 256, 0, stream>>>(y4, fyg, fyb, hyln);
  G(hyln, fyw1T, hy16, fyb1, nullptr, DIM_, 0, 0, DIM_, 0, 0,
    FFN_, 0, 0, (int)MY, FFN_, DIM_, 1, 1, 1 | 2 | 8, 1.f);
  G(hy16, fyw2T, y5, fyb2, y4, FFN_, 0, 0, FFN_, 0, 0,
    DIM_, 0, 0, (int)MY, DIM_, FFN_, 1, 1, 1 | 4, 1.f);
}
